// ChittaEncoder_17918603559310
// MI455X (gfx1250) — matrix operand order repaired and checked by reading, not compiled, not run on hardware
//
#include <hip/hip_runtime.h>
#include <stdint.h>

// ---------------------------------------------------------------------------
// ChittaEncoder for MI455X (gfx1250, wave32, WMMA)
//
//   P = (seeds @ W) / sqrt(128)            (tiny one-off kernel, f32->f16)
//   scores = x @ P^T                       (v_wmma_f32_16x16x32_f16, P in LDS)
//   top-4 via sortable-key min/max chains  (no cndmask; index packed in key)
//   2x software-pipelined N-tile loop      (top-k co-executes with WMMA chain)
//   softmax + weighted seed gather         (float4 coalesced)
// ---------------------------------------------------------------------------

typedef __attribute__((ext_vector_type(16))) _Float16 v16h;
typedef __attribute__((ext_vector_type(8)))  float    v8f;

#define D_MODEL       128
#define N_SEEDS       500
#define N_PAD         512              // padded seed count (32 N-tiles of 16)
#define N_TILES       (N_PAD / 16)
#define WAVES_PER_WG  8
#define THREADS       (WAVES_PER_WG * 32)
#define ITERS_PER_WG  4                // M-tiles per wave per workgroup
#define MTILES_PER_WG (ITERS_PER_WG * WAVES_PER_WG)

static __device__ __forceinline__ uint32_t umax32(uint32_t a, uint32_t b) { return a > b ? a : b; }
static __device__ __forceinline__ uint32_t umin32(uint32_t a, uint32_t b) { return a < b ? a : b; }

// Insert `key` into descending sorted top-4 (k0>=k1>=k2>=k3).
// Pure v_max_u32/v_min_u32 chain: 7 VALU, no compares, no cndmask.
static __device__ __forceinline__ void topk_insert(uint32_t& k0, uint32_t& k1,
                                                   uint32_t& k2, uint32_t& k3,
                                                   uint32_t key) {
    uint32_t m, n = key;
    m = umax32(k0, n); n = umin32(k0, n); k0 = m;
    m = umax32(k1, n); n = umin32(k1, n); k1 = m;
    m = umax32(k2, n); n = umin32(k2, n); k2 = m;
    k3 = umax32(k3, n);
}

// Fold one 16x16 f32 score tile into the per-lane top-4 key state.
// colk = (t<<4) | lane_column; keys pack score (23 MSB-equivalent bits,
// monotonic transform) + column index (9 LSBs).
static __device__ __forceinline__ void topk_update_tile(const v8f& c, uint32_t colk,
                                                        uint32_t tk[8][4]) {
    const bool valid = colk < N_SEEDS;
#pragma unroll
    for (int r = 0; r < 8; ++r) {
        uint32_t u   = __float_as_uint(c[r]);
        uint32_t key = u ^ (uint32_t)(((int32_t)u >> 31) | 0x80000000);
        key = (key & 0xFFFFFE00u) | colk;
        key = valid ? key : 0u;
        topk_insert(tk[r][0], tk[r][1], tk[r][2], tk[r][3], key);
    }
}

// ---------------------------------------------------------------------------
// Kernel 1: P[j][e] = (sum_d seeds[j][d] * W[d][e]) * rsqrt(128) as f16,
// zero-padded to 512 rows. 512 blocks x 128 threads; W reads coalesced,
// seeds reads wave-uniform (scalar cached).
// ---------------------------------------------------------------------------
__global__ void fold_proj_kernel(const float* __restrict__ W,
                                 const float* __restrict__ seeds,
                                 _Float16* __restrict__ P) {
    const int j = blockIdx.x;
    const int e = threadIdx.x;
    float acc = 0.f;
    if (j < N_SEEDS) {
#pragma unroll 8
        for (int d = 0; d < D_MODEL; ++d)
            acc += seeds[j * D_MODEL + d] * W[d * D_MODEL + e];
        acc *= 0.08838834764831845f;   // 1/sqrt(128) folded into scores
    }
    P[j * D_MODEL + e] = (_Float16)acc;
}

// ---------------------------------------------------------------------------
// Kernel 2: per 16-row M-tile (one wave each):
//   - A fragments (x rows, f32->f16) held in VGPRs, reused over 32 N-tiles
//   - B fragments streamed from LDS-resident P (one v16h ds load per frag/lane)
//   - 4 WMMAs per N-tile accumulate a 16x16 f32 score tile; two tiles in
//     flight so top-k VALU co-executes with the next WMMA chain
//   - shfl_xor butterfly merge -> softmax -> seeds gather -> field/attn store
// ---------------------------------------------------------------------------
__global__ __launch_bounds__(THREADS)
void score_topk_kernel(const float* __restrict__ x,
                       const float* __restrict__ seeds,
                       const _Float16* __restrict__ Pg,
                       float* __restrict__ field,
                       float* __restrict__ attn_out,
                       int mtiles) {
    extern __shared__ __align__(32) _Float16 sP[];   // 512 x 128 f16 = 128 KB

    // ---- cooperative fill of LDS with P (one-off per workgroup) ----
    {
        const uint4* src = (const uint4*)Pg;
        uint4*       dst = (uint4*)sP;
        const int total = (N_PAD * D_MODEL * 2) / 16;   // 8192 uint4
        for (int i = threadIdx.x; i < total; i += THREADS) dst[i] = src[i];
    }
    __syncthreads();

    const int wave = threadIdx.x >> 5;
    const int lane = threadIdx.x & 31;
    const int lj   = lane & 15;        // column-class / A-row selector
    const int hk   = lane >> 4;        // K-half selector (0 or 1)

    for (int it = 0; it < ITERS_PER_WG; ++it) {
        const int mtile = (blockIdx.x * ITERS_PER_WG + it) * WAVES_PER_WG + wave;
        if (mtile >= mtiles) break;              // wave-uniform: EXEC stays full
        const int rbase = mtile * 16;

        // ---- A fragments: 16x32 f16, contiguous K order (B uses the same):
        // lane l (both halves share row M=lj): K = k0+hk*16 .. +15
        const float4* xr4 = (const float4*)(x + (size_t)(rbase + lj) * D_MODEL);
        v16h a0, a1, a2, a3;
        {
#define MAKE_AFRAG(dst, kc)                                                    \
            {                                                                  \
                const int i0 = (kc) * 8 + hk * 4;                              \
                float4 f0 = xr4[i0], f1 = xr4[i0 + 1];                         \
                float4 f2 = xr4[i0 + 2], f3 = xr4[i0 + 3];                     \
                dst[0] = (_Float16)f0.x;  dst[1] = (_Float16)f0.y;             \
                dst[2] = (_Float16)f0.z;  dst[3] = (_Float16)f0.w;             \
                dst[4] = (_Float16)f1.x;  dst[5] = (_Float16)f1.y;             \
                dst[6] = (_Float16)f1.z;  dst[7] = (_Float16)f1.w;             \
                dst[8]  = (_Float16)f2.x; dst[9]  = (_Float16)f2.y;            \
                dst[10] = (_Float16)f2.z; dst[11] = (_Float16)f2.w;            \
                dst[12] = (_Float16)f3.x; dst[13] = (_Float16)f3.y;            \
                dst[14] = (_Float16)f3.z; dst[15] = (_Float16)f3.w;            \
            }
            MAKE_AFRAG(a0, 0) MAKE_AFRAG(a1, 1) MAKE_AFRAG(a2, 2) MAKE_AFRAG(a3, 3)
#undef MAKE_AFRAG
        }

        // ---- prefetch next M-tile's x rows behind the long N-tile sweep ----
        // (global_prefetch_b8: 4 cachelines covering this lane's next 512B row)
        {
            const int nmt = (blockIdx.x * ITERS_PER_WG + it + 1) * WAVES_PER_WG + wave;
            if (it + 1 < ITERS_PER_WG && nmt < mtiles) {
                const float* nx = x + (size_t)(nmt * 16 + lj) * D_MODEL;
                __builtin_prefetch(nx,      0, 1);
                __builtin_prefetch(nx + 32, 0, 1);
                __builtin_prefetch(nx + 64, 0, 1);
                __builtin_prefetch(nx + 96, 0, 1);
            }
        }

        // ---- running top-4 sortable keys: 8 rows x 4 per lane ----
        uint32_t tk[8][4];
#pragma unroll
        for (int r = 0; r < 8; ++r) { tk[r][0] = tk[r][1] = tk[r][2] = tk[r][3] = 0u; }

        const v8f czero = {};           // persistent zero C operand

        // ---- stream 32 N-tiles, two independent accumulator chains so the
        //      top-k VALU of tile t overlaps the WMMA chain of tile t+1 ----
        for (int t = 0; t < N_TILES; t += 2) {
            const _Float16* bpA = sP + ((t * 16 + lj) * D_MODEL) + hk * 16;
            const _Float16* bpB = bpA + 16 * D_MODEL;

            v16h bA0 = *(const v16h*)(bpA);
            v16h bA1 = *(const v16h*)(bpA + 32);
            v16h bA2 = *(const v16h*)(bpA + 64);
            v16h bA3 = *(const v16h*)(bpA + 96);
            v8f cA;
            cA = __builtin_amdgcn_wmma_f32_16x16x32_f16(false, a0, false, bA0, (short)0, czero, false, false);
            cA = __builtin_amdgcn_wmma_f32_16x16x32_f16(false, a1, false, bA1, (short)0, cA,    false, false);
            cA = __builtin_amdgcn_wmma_f32_16x16x32_f16(false, a2, false, bA2, (short)0, cA,    false, false);
            cA = __builtin_amdgcn_wmma_f32_16x16x32_f16(false, a3, false, bA3, (short)0, cA,    false, false);

            v16h bB0 = *(const v16h*)(bpB);
            v16h bB1 = *(const v16h*)(bpB + 32);
            v16h bB2 = *(const v16h*)(bpB + 64);
            v16h bB3 = *(const v16h*)(bpB + 96);
            v8f cB;
            cB = __builtin_amdgcn_wmma_f32_16x16x32_f16(false, a0, false, bB0, (short)0, czero, false, false);
            cB = __builtin_amdgcn_wmma_f32_16x16x32_f16(false, a1, false, bB1, (short)0, cB,    false, false);

            // top-k for tile t co-executes with tile t+1's WMMA chain
            topk_update_tile(cA, ((uint32_t)t << 4) | (uint32_t)lj, tk);

            cB = __builtin_amdgcn_wmma_f32_16x16x32_f16(false, a2, false, bB2, (short)0, cB,    false, false);
            cB = __builtin_amdgcn_wmma_f32_16x16x32_f16(false, a3, false, bB3, (short)0, cB,    false, false);

            topk_update_tile(cB, ((uint32_t)(t + 1) << 4) | (uint32_t)lj, tk);
        }

        // ---- butterfly merge of the 16 column-classes (lanes 0-15 / 16-31) ----
#pragma unroll
        for (int r = 0; r < 8; ++r) {
#pragma unroll
            for (int stage = 0; stage < 4; ++stage) {
                const int msk = 1 << stage;
                uint32_t p0 = (uint32_t)__shfl_xor((int)tk[r][0], msk, 32);
                uint32_t p1 = (uint32_t)__shfl_xor((int)tk[r][1], msk, 32);
                uint32_t p2 = (uint32_t)__shfl_xor((int)tk[r][2], msk, 32);
                uint32_t p3 = (uint32_t)__shfl_xor((int)tk[r][3], msk, 32);
                topk_insert(tk[r][0], tk[r][1], tk[r][2], tk[r][3], p0);
                topk_insert(tk[r][0], tk[r][1], tk[r][2], tk[r][3], p1);
                topk_insert(tk[r][0], tk[r][1], tk[r][2], tk[r][3], p2);
                topk_insert(tk[r][0], tk[r][1], tk[r][2], tk[r][3], p3);
            }
        }

        // ---- decode, softmax, gather, store ----
#pragma unroll
        for (int r = 0; r < 8; ++r) {
            const int row = rbase + r + hk * 8;     // C layout: lanes>=16 hold M=r+8
            float sc[4]; int col[4];
#pragma unroll
            for (int i = 0; i < 4; ++i) {
                const uint32_t kk = tk[r][i];
                col[i] = (int)(kk & 511u);
                const uint32_t vb = kk & 0xFFFFFE00u;
                const uint32_t uu = (kk & 0x80000000u) ? (vb ^ 0x80000000u) : ~vb;
                sc[i] = __uint_as_float(uu);        // |rel err| <= 2^-14
            }
            const float e1 = __expf(sc[1] - sc[0]);
            const float e2 = __expf(sc[2] - sc[0]);
            const float e3 = __expf(sc[3] - sc[0]);
            const float inv = 1.0f / (1.0f + e1 + e2 + e3);
            float aw[4] = { inv, e1 * inv, e2 * inv, e3 * inv };

            const int dbase = lj * 8;               // each lane owns 8 of 128 dims
            float c0 = 0, c1 = 0, c2 = 0, c3 = 0, c4 = 0, c5 = 0, c6 = 0, c7 = 0;
#pragma unroll
            for (int i = 0; i < 4; ++i) {
                const float4* sr = (const float4*)(seeds + (size_t)col[i] * D_MODEL + dbase);
                float4 s0 = sr[0], s1 = sr[1];
                c0 += aw[i] * s0.x; c1 += aw[i] * s0.y; c2 += aw[i] * s0.z; c3 += aw[i] * s0.w;
                c4 += aw[i] * s1.x; c5 += aw[i] * s1.y; c6 += aw[i] * s1.z; c7 += aw[i] * s1.w;
            }
            float4* op = (float4*)(field + (size_t)row * D_MODEL + dbase);
            op[0] = make_float4(c0, c1, c2, c3);
            op[1] = make_float4(c4, c5, c6, c7);
            if (lj == 0) {
                *(float4*)(attn_out + (size_t)row * 4) = make_float4(aw[0], aw[1], aw[2], aw[3]);
            }
        }
    }
}

// ---------------------------------------------------------------------------
extern "C" void kernel_launch(void* const* d_in, const int* in_sizes, int n_in,
                              void* d_out, int out_size, void* d_ws, size_t ws_size,
                              hipStream_t stream) {
    const float* x     = (const float*)d_in[0];   // [B, 128]
    const float* W     = (const float*)d_in[1];   // [128, 128]
    const float* seeds = (const float*)d_in[2];   // [500, 128]
    float*       out   = (float*)d_out;           // field [B,128] ++ attn [B,4]

    const int B = in_sizes[0] / D_MODEL;
    _Float16* P = (_Float16*)d_ws;                // 512*128 f16 = 128 KB scratch

    // 1) fold W into seed projections (writes all 512 padded rows)
    fold_proj_kernel<<<N_PAD, D_MODEL, 0, stream>>>(W, seeds, P);

    // 2) WMMA scores + fused top-k / softmax / gather
    const int mtiles = B / 16;
    const int wgs = (mtiles + MTILES_PER_WG - 1) / MTILES_PER_WG;
    const size_t lds_bytes = (size_t)N_PAD * D_MODEL * sizeof(_Float16);
    score_topk_kernel<<<wgs, THREADS, lds_bytes, stream>>>(
        x, seeds, P, out, out + (size_t)B * D_MODEL, mtiles);
}
